// SinkhornDistance_POT_53206054863638
// MI455X (gfx1250) — matrix operand order repaired and checked by reading, not compiled, not run on hardware
//
#include <hip/hip_runtime.h>
#include <math.h>

#define P 4096
#define Dd 512
#define LOGP 8.317766166719343f   // log(4096); eps = 1
#define TK 32
#define NT (Dd / TK)              // 16 k-tiles

typedef __attribute__((ext_vector_type(16))) _Float16 v16h;
typedef __attribute__((ext_vector_type(8)))  float    v8f;

union HFrag { v16h h; float4 f[2]; };

// gfx1250 async copy, GVS mode: mem_addr = SGPR base + VGPR 32-bit offset + imm offset.
// 16B per lane into LDS, tracked by ASYNCcnt.
__device__ __forceinline__ void async_ld_b128(unsigned lds_addr, unsigned voff,
                                              const void* sbase, int ioff) {
  asm volatile("global_load_async_to_lds_b128 %0, %1, %2 offset:%3"
               :: "v"(lds_addr), "v"(voff), "s"(sbase), "n"(ioff)
               : "memory");
}
__device__ __forceinline__ void wait_asynccnt0() {
  asm volatile("s_wait_asynccnt 0x0" ::: "memory");
}
__device__ __forceinline__ unsigned lds_off(const void* p) {
  return (unsigned)(size_t)p;   // LDS aperture: low 32 bits of generic addr == LDS address
}

// ---------------------------------------------------------------- prep: f16 copies + squared norms
__global__ __launch_bounds__(256) void prep_kernel(const float* __restrict__ x,
                                                   const float* __restrict__ y,
                                                   _Float16* __restrict__ xh,
                                                   _Float16* __restrict__ yh,
                                                   float* __restrict__ x2,
                                                   float* __restrict__ y2) {
  const int b = blockIdx.x;
  const int t = threadIdx.x;
  const float* src;
  _Float16* dsth;
  float* dst2;
  if (b < P) { src = x + (size_t)b * Dd; dsth = xh + (size_t)b * Dd; dst2 = x2 + b; }
  else       { int r = b - P; src = y + (size_t)r * Dd; dsth = yh + (size_t)r * Dd; dst2 = y2 + r; }
  float s = 0.f;
  for (int d = t; d < Dd; d += 256) {
    float v = src[d];
    dsth[d] = (_Float16)v;
    s += v * v;
  }
  __shared__ float sm[256];
  sm[t] = s; __syncthreads();
  for (int off = 128; off > 0; off >>= 1) {
    if (t < off) sm[t] += sm[t + off];
    __syncthreads();
  }
  if (t == 0) *dst2 = sm[0];
}

__global__ void zero_fg_kernel(float* __restrict__ f, float* __restrict__ g) {
  int j = blockIdx.x * 256 + threadIdx.x;
  if (j < P) { f[j] = 0.f; g[j] = 0.f; }
}

// ---------------------------------------------------------------- GEMM: C = max(x2 + y2 - 2*x.y, 0)
// 128x128 tile per 256-thread block (8 waves). Double-buffered LDS tiles filled by
// global_load_async_to_lds_b128 (GVS addressing); tile k+1 DMA overlaps the 8 WMMAs of tile k.
__global__ __launch_bounds__(256) void gemm_cost_kernel(const _Float16* __restrict__ xh,
                                                        const _Float16* __restrict__ yh,
                                                        const float* __restrict__ x2,
                                                        const float* __restrict__ y2,
                                                        float* __restrict__ C) {
  __shared__ _Float16 As[2][128 * 40];   // [row][k] pitch 40 halves (80B, 16B aligned)
  __shared__ _Float16 Bs[2][128 * 40];   // [col][k]
  const int t    = threadIdx.x;
  const int lane = t & 31;
  const int w    = t >> 5;
  const int i0   = blockIdx.y * 128;
  const int j0   = blockIdx.x * 128;

  v8f acc[8] = {};

  // loader role: thread t copies 32B (two b128) per matrix per tile
  const int lr = t >> 1;            // row/col 0..127
  const int lo = (t & 1) * 16;      // 16-half (32B) chunk
  const unsigned gxo = (unsigned)(((i0 + lr) * Dd + lo) * 2);   // byte offset into xh
  const unsigned gyo = (unsigned)(((j0 + lr) * Dd + lo) * 2);   // byte offset into yh
  const unsigned la0 = lds_off(&As[0][lr * 40 + lo]);
  const unsigned la1 = lds_off(&As[1][lr * 40 + lo]);
  const unsigned lb0 = lds_off(&Bs[0][lr * 40 + lo]);
  const unsigned lb1 = lds_off(&Bs[1][lr * 40 + lo]);

  // fragment addressing (per WMMA layout)
  const int arow = w * 16 + (lane & 15);
  const int abk  = (lane >> 4) * 8;     // A: halves [0..7]=K+0..7, [8..15]=K+16..23
  const int bk   = (lane >> 4) * 8;     // B: halves [0..7]=K+0..7, [8..15]=K+16..23

  // prologue: tile 0 -> buffer 0 (tile stride = TK halves = 64 bytes, folded into imm offset)
  async_ld_b128(la0,      gxo, xh, 0);
  async_ld_b128(la0 + 16, gxo, xh, 16);
  async_ld_b128(lb0,      gyo, yh, 0);
  async_ld_b128(lb0 + 16, gyo, yh, 16);

#pragma unroll
  for (int kt = 0; kt < NT; ++kt) {
    const int buf = kt & 1;
    wait_asynccnt0();          // this wave's copies into `buf` done
    __syncthreads();           // everyone's copies visible

    if (kt + 1 < NT) {         // kick off tile kt+1 into the other buffer
      const int go = (kt + 1) * (TK * 2);   // compile-time constant after unroll
      const unsigned na = buf ? la0 : la1;
      const unsigned nb = buf ? lb0 : lb1;
      async_ld_b128(na,      gxo, xh, go);
      async_ld_b128(na + 16, gxo, xh, go + 16);
      async_ld_b128(nb,      gyo, yh, go);
      async_ld_b128(nb + 16, gyo, yh, go + 16);
    }

    const _Float16* Ab = &As[buf][0];
    const _Float16* Bb = &Bs[buf][0];

    HFrag fa;
    fa.f[0] = *(const float4*)(Ab + arow * 40 + abk);
    fa.f[1] = *(const float4*)(Ab + arow * 40 + abk + 16);

    HFrag fb[8];
#pragma unroll
    for (int n = 0; n < 8; ++n) {
      const int bcol = n * 16 + (lane & 15);
      fb[n].f[0] = *(const float4*)(Bb + bcol * 40 + bk);
      fb[n].f[1] = *(const float4*)(Bb + bcol * 40 + bk + 16);
    }
#pragma unroll
    for (int n = 0; n < 8; ++n)
      acc[n] = __builtin_amdgcn_wmma_f32_16x16x32_f16(
          false, fa.h, false, fb[n].h, (short)0, acc[n], false, false);

    __syncthreads();           // done reading `buf` before it is refilled
  }

  // D layout: element v of v8f -> M = v + 8*(lane>=16); N = lane&15
  const int mrow  = i0 + w * 16 + ((lane >> 4) << 3);
  const int ncol0 = j0 + (lane & 15);
#pragma unroll
  for (int n = 0; n < 8; ++n) {
    const int col = ncol0 + n * 16;
    const float yv = y2[col];
#pragma unroll
    for (int v = 0; v < 8; ++v) {
      const int row = mrow + v;
      float c = x2[row] + yv - 2.0f * acc[n][v];
      C[(size_t)row * P + col] = fmaxf(c, 0.0f);
    }
  }
}

// ---------------------------------------------------------------- Sinkhorn row pass: f = -logP - LSE_j(g[j]-C[i,j])
__global__ __launch_bounds__(256) void row_pass_kernel(const float* __restrict__ C,
                                                       const float* __restrict__ g,
                                                       float* __restrict__ f) {
  const int i = blockIdx.x;
  const int t = threadIdx.x;
  const float* Crow = C + (size_t)i * P;
  float m = -INFINITY, s = 0.f;
  for (int j = t; j < P; j += 256) {
    float v = g[j] - Crow[j];
    float nm = fmaxf(m, v);
    s = s * __expf(m - nm) + __expf(v - nm);
    m = nm;
  }
  __shared__ float sm[256], ss[256];
  sm[t] = m; ss[t] = s; __syncthreads();
  for (int off = 128; off > 0; off >>= 1) {
    if (t < off) {
      float m2 = sm[t + off], s2 = ss[t + off];
      float M = fmaxf(sm[t], m2);
      ss[t] = ss[t] * __expf(sm[t] - M) + s2 * __expf(m2 - M);
      sm[t] = M;
    }
    __syncthreads();
  }
  if (t == 0) f[i] = -LOGP - (sm[0] + __logf(ss[0]));
}

// ---------------------------------------------------------------- col pass (coalesced partials):
// block = 64 columns x 512 rows; warp reads contiguous 128B row segments.
__global__ __launch_bounds__(256) void col_partial_kernel(const float* __restrict__ C,
                                                          const float* __restrict__ f,
                                                          float* __restrict__ pm,
                                                          float* __restrict__ ps) {
  const int t  = threadIdx.x;
  const int cb = blockIdx.x & 63;    // column block (64 cols)
  const int rs = blockIdx.x >> 6;    // row slice (512 rows), 8 slices
  const int c  = cb * 64 + (t & 63);
  const int rl = t >> 6;             // 4 row lanes per column
  float m = -INFINITY, s = 0.f;
  const int rbase = rs * 512;
  for (int i = rbase + rl; i < rbase + 512; i += 4) {
    float v = f[i] - C[(size_t)i * P + c];
    float nm = fmaxf(m, v);
    s = s * __expf(m - nm) + __expf(v - nm);
    m = nm;
  }
  __shared__ float sm[256], ss[256];
  sm[t] = m; ss[t] = s; __syncthreads();
  if (rl < 2) {
    float m2 = sm[t + 128], s2 = ss[t + 128];
    float M = fmaxf(sm[t], m2);
    ss[t] = ss[t] * __expf(sm[t] - M) + s2 * __expf(m2 - M);
    sm[t] = M;
  }
  __syncthreads();
  if (rl == 0) {
    float m2 = sm[t + 64], s2 = ss[t + 64];
    float M = fmaxf(sm[t], m2);
    float S = ss[t] * __expf(sm[t] - M) + s2 * __expf(m2 - M);
    pm[(size_t)rs * P + c] = M;
    ps[(size_t)rs * P + c] = S;
  }
}

__global__ void col_combine_kernel(const float* __restrict__ pm,
                                   const float* __restrict__ ps,
                                   float* __restrict__ g) {
  const int j = blockIdx.x * 256 + threadIdx.x;
  if (j >= P) return;
  float m = -INFINITY, s = 0.f;
#pragma unroll
  for (int rs = 0; rs < 8; ++rs) {
    float m2 = pm[(size_t)rs * P + j], s2 = ps[(size_t)rs * P + j];
    float M = fmaxf(m, m2);
    s = s * __expf(m - M) + s2 * __expf(m2 - M);
    m = M;
  }
  g[j] = -LOGP - (m + __logf(s));
}

// ---------------------------------------------------------------- finalize: pi = exp(f+g-C), cost = sum(pi*C)
__global__ __launch_bounds__(256) void pi_cost_kernel(const float* __restrict__ C,
                                                      const float* __restrict__ f,
                                                      const float* __restrict__ g,
                                                      float* __restrict__ PI,
                                                      float* __restrict__ part) {
  const int i = blockIdx.x;
  const int t = threadIdx.x;
  const float fi = f[i];
  const float* Crow = C + (size_t)i * P;
  float* Prow = PI + (size_t)i * P;
  float acc = 0.f;
  for (int j = t; j < P; j += 256) {
    float c = Crow[j];
    float p = __expf(fi + g[j] - c);
    __builtin_nontemporal_store(p, &Prow[j]);   // write-once stream, keep C resident in L2
    acc += p * c;
  }
  __shared__ float sm[256];
  sm[t] = acc; __syncthreads();
  for (int off = 128; off > 0; off >>= 1) {
    if (t < off) sm[t] += sm[t + off];
    __syncthreads();
  }
  if (t == 0) part[i] = sm[0];
}

__global__ void cost_final_kernel(const float* __restrict__ part, float* __restrict__ out) {
  const int t = threadIdx.x;
  float s = 0.f;
  for (int i = t; i < P; i += 256) s += part[i];
  __shared__ float sm[256];
  sm[t] = s; __syncthreads();
  for (int off = 128; off > 0; off >>= 1) {
    if (t < off) sm[t] += sm[t + off];
    __syncthreads();
  }
  if (t == 0) out[0] = sm[0];
}

// ----------------------------------------------------------------
extern "C" void kernel_launch(void* const* d_in, const int* in_sizes, int n_in,
                              void* d_out, int out_size, void* d_ws, size_t ws_size,
                              hipStream_t stream) {
  (void)in_sizes; (void)n_in; (void)out_size; (void)ws_size;
  const float* x = (const float*)d_in[0];
  const float* y = (const float*)d_in[1];

  float* out  = (float*)d_out;
  float* cost = out;                          // [1]
  float* PI   = out + 1;                      // [4096*4096]
  float* C    = out + 1 + (size_t)P * P;      // [4096*4096]

  // workspace layout (~8.4 MB)
  _Float16* xh = (_Float16*)d_ws;             // 4 MB
  _Float16* yh = xh + (size_t)P * Dd;         // 4 MB
  float* wsf  = (float*)(yh + (size_t)P * Dd);
  float* x2   = wsf;                          // 4096
  float* y2   = x2 + P;                       // 4096
  float* f    = y2 + P;                       // 4096
  float* g    = f + P;                        // 4096
  float* pm   = g + P;                        // 8*4096
  float* ps   = pm + 8 * P;                   // 8*4096
  float* part = ps + 8 * P;                   // 4096

  prep_kernel<<<2 * P, 256, 0, stream>>>(x, y, xh, yh, x2, y2);
  zero_fg_kernel<<<P / 256, 256, 0, stream>>>(f, g);
  gemm_cost_kernel<<<dim3(P / 128, P / 128), 256, 0, stream>>>(xh, yh, x2, y2, C);

  for (int it = 0; it < 100; ++it) {
    row_pass_kernel<<<P, 256, 0, stream>>>(C, g, f);
    col_partial_kernel<<<512, 256, 0, stream>>>(C, f, pm, ps);
    col_combine_kernel<<<P / 256, 256, 0, stream>>>(pm, ps, g);
  }

  pi_cost_kernel<<<P, 256, 0, stream>>>(C, f, g, PI, part);
  cost_final_kernel<<<1, 256, 0, stream>>>(part, cost);
}
